// MambaIntegerBlock_901943132627
// MI455X (gfx1250) — matrix operand order repaired and checked by reading, not compiled, not run on hardware
//
#include <hip/hip_runtime.h>
#include <hip/hip_bf16.h>
#include <stdint.h>

#define DMODEL 1024
#define DINNER 2048
#define LSEQ   2048
#define NBATCH 4
#define NTOK   (NBATCH*LSEQ)   /* 8192 tokens */
#define DTRANK 64
#define XZCOLS (2*DINNER)      /* 4096 */

typedef __attribute__((ext_vector_type(16))) _Float16 v16h;
typedef __attribute__((ext_vector_type(8)))  _Float16 v8h;
typedef __attribute__((ext_vector_type(8)))  float    v8f;
typedef __attribute__((ext_vector_type(4)))  unsigned int u32x4;
typedef __attribute__((ext_vector_type(8)))  int          i32x8;
typedef __attribute__((ext_vector_type(4)))  int          i32x4;

// ---------------------------------------------------------------- helpers ----

__device__ __forceinline__ float pow2scale(float var) {
  float v = var + 1e-9f;
  float s = 1.0f;
  if (v >= 4.0f)      s = 0.5f;
  if (v >= 16.0f)     s = 0.25f;
  if (v >= 64.0f)     s = 0.125f;
  if (v >= 256.0f)    s = 0.0625f;
  if (v >= 1024.0f)   s = 0.03125f;
  if (v >= 4096.0f)   s = 0.015625f;
  if (v >= 16384.0f)  s = 0.0078125f;
  if (v >= 65536.0f)  s = 0.00390625f;
  if (v < 1.0f)    s = 1.0f;
  if (v < 0.25f)   s = 2.0f;
  if (v < 0.0625f) s = 4.0f;
  return s;
}

__device__ __forceinline__ float rsqrt_newton3(float y) {
  y = fminf(fmaxf(y, 1e-6f), 1e6f);
  float r = (y > 16.0f) ? 0.125f : 0.5f;
  if (y > 64.0f)  r = 0.0625f;
  if (y > 256.0f) r = 0.03125f;
  if (y < 4.0f)   r = 1.0f;
  if (y < 1.0f)   r = 2.0f;
  if (y < 0.25f)  r = 4.0f;
  #pragma unroll
  for (int i = 0; i < 3; ++i)
    r = fminf(fmaxf(r * (1.5f - 0.5f * y * r * r), 1e-6f), 1e3f);
  return r;
}

__device__ __forceinline__ float squareplus(float x) {
  float y = fminf(fmaxf(x * x + 4.0f, 1e-6f), 1e6f);
  float s = fminf(fmaxf(y * rsqrt_newton3(y), 0.0f), 1e3f);
  return 0.5f * (x + s);
}

// A 16x32 f16 fragment (ISA 7.12.2): lane<16 -> row=lane, K = k0..k0+7 | k0+16..k0+23
//                                    lane>=16 -> row=lane-16, K = k0+8..15 | k0+24..31
__device__ __forceinline__ v16h load_frag_a(const _Float16* __restrict__ A, int lda,
                                            int row0, int k0, int lane) {
  const _Float16* p = A + (long)(row0 + (lane & 15)) * lda + k0 + ((lane & 16) ? 8 : 0);
  union { v16h v; v8h h[2]; } u;
  u.h[0] = *(const v8h*)(p);
  u.h[1] = *(const v8h*)(p + 16);
  return u.v;
}
// B 32x16 f16 (KxN): lane<16 -> n=lane, K=k0..k0+7 | k0+16..k0+23 ; lane>=16 -> n=lane-16, K=k0+8..15 | k0+24..31
// p points at (col row base, k0)
__device__ __forceinline__ v16h load_frag_b_ptr(const _Float16* p) {
  union { v16h v; v8h h[2]; } u;
  u.h[0] = *(const v8h*)(p);
  u.h[1] = *(const v8h*)(p + 16);
  return u.v;
}

#define WMMA_F16(a, b, c) \
  __builtin_amdgcn_wmma_f32_16x16x32_f16(false, (a), false, (b), (short)0, (c), false, false)

// -------------------------------------------------- TDM: stage B tile to LDS --
// 2-D tile descriptor (ISA 8.3/8.4): 32 rows x 64 halves (128B/row, compact in LDS).
__device__ __forceinline__ void tdm_issue_b(const _Float16* Bw, long row0, int kc,
                                            int K, int Nrows, unsigned lds_byte) {
  unsigned long long ga = (unsigned long long)(size_t)(Bw + row0 * (long)K + kc);
  u32x4 g0;
  g0[0] = 1u;                                          // count=1 (valid user D#)
  g0[1] = lds_byte;                                    // lds_addr (bytes)
  g0[2] = (unsigned)(ga & 0xffffffffull);              // global_addr[31:0]
  g0[3] = (unsigned)((ga >> 32) & 0x1ffffffull)        // global_addr[56:32]
        | (2u << 30);                                  // type=2 ("image")
  i32x8 g1;
  g1[0] = (int)(1u << 16);                             // data_size=1 -> 2 bytes
  g1[1] = (int)((((unsigned)K) & 0xffffu) << 16);      // tensor_dim0[15:0]
  g1[2] = (int)(((unsigned)K >> 16) & 0xffffu) |       // tensor_dim0[31:16]
          (int)(((unsigned)Nrows & 0xffffu) << 16);    // tensor_dim1[15:0]
  g1[3] = (int)(((unsigned)Nrows >> 16) & 0xffffu) |   // tensor_dim1[31:16]
          (int)(64u << 16);                            // tile_dim0 = 64 halves
  g1[4] = 32;                                          // tile_dim1 = 32 rows, tile_dim2=0
  g1[5] = K;                                           // tensor_dim0_stride[31:0]
  g1[6] = 0;                                           // stride hi, dim1_stride lo
  g1[7] = 0;
  i32x4 z4 = {0, 0, 0, 0};                             // groups 2/3 unused (<=2D)
  i32x8 z8 = {0, 0, 0, 0, 0, 0, 0, 0};
  __builtin_amdgcn_tensor_load_to_lds(g0, g1, z4, z4, z8, 0);  // 6-arg toolchain form
}

// -------------------------------------------------- big GEMM (TDM + WMMA) ----
// 128 threads = 4 wave32. Block tile 64(M) x 128(N); wave tile 32x64 (2x4 frags).
// B panel double-buffered in LDS via tensor_load_to_lds; A direct from global.
// EPI: 0 = plain f32 store, 1 = decay epilogue (aux=b_dt), 2 = residual (aux).
template <int EPI>
__global__ __launch_bounds__(128) void gemm_tdm_kernel(const _Float16* __restrict__ A,
                                                       const _Float16* __restrict__ Bw,
                                                       const float* __restrict__ aux,
                                                       float* __restrict__ C,
                                                       int N, int K) {
  __shared__ _Float16 Bsm[2 * 128 * 64];   // 2 x 16KB panels: [row 0..127][k 0..63]
  const int lane = threadIdx.x & 31, w = threadIdx.x >> 5;
  const int m0 = blockIdx.y * 64 + (w & 1) * 32;
  const int n0b = blockIdx.x * 128;        // block N origin
  const int n0w = (w >> 1) * 64;           // wave-local N origin inside panel
  const unsigned lds_base = (unsigned)(unsigned long long)(size_t)(void*)Bsm;

  v8f c[2][4] = {};

  // prologue: each wave DMAs its 32-row slice of panel 0
  tdm_issue_b(Bw, (long)n0b + w * 32, 0, K, N, lds_base + (unsigned)(w * 32 * 128));
  __builtin_amdgcn_s_wait_tensorcnt((short)0);
  __syncthreads();

  for (int kc = 0; kc < K; kc += 64) {
    const int cur = (kc >> 6) & 1;
    if (kc + 64 < K)  // overlap next panel DMA with compute
      tdm_issue_b(Bw, (long)n0b + w * 32, kc + 64, K, N,
                  lds_base + (unsigned)((cur ^ 1) * 16384 + w * 32 * 128));

    const _Float16* Bp = Bsm + cur * (128 * 64);
    #pragma unroll
    for (int ks = 0; ks < 2; ++ks) {
      const int kg = kc + ks * 32;
      if (kg + 32 < K) {
        __builtin_prefetch(A + (long)(m0 + (lane & 15)) * K + kg + 32, 0, 1);
      }
      v16h a0 = load_frag_a(A, K, m0,      kg, lane);
      v16h a1 = load_frag_a(A, K, m0 + 16, kg, lane);
      v16h b[4];
      #pragma unroll
      for (int fn = 0; fn < 4; ++fn)
        b[fn] = load_frag_b_ptr(Bp + (n0w + fn * 16 + (lane & 15)) * 64 +
                                ks * 32 + ((lane & 16) ? 8 : 0));
      #pragma unroll
      for (int fn = 0; fn < 4; ++fn) {
        c[0][fn] = WMMA_F16(a0, b[fn], c[0][fn]);
        c[1][fn] = WMMA_F16(a1, b[fn], c[1][fn]);
      }
    }
    if (kc + 64 < K) __builtin_amdgcn_s_wait_tensorcnt((short)0);
    __syncthreads();
  }

  const int rb = (lane & 16) ? 8 : 0, nc = lane & 15;
  #pragma unroll
  for (int fm = 0; fm < 2; ++fm)
    #pragma unroll
    for (int fn = 0; fn < 4; ++fn)
      #pragma unroll
      for (int r = 0; r < 8; ++r) {
        const int row = m0 + fm * 16 + rb + r;
        const int n   = n0b + n0w + fn * 16 + nc;
        const long idx = (long)row * N + n;
        if constexpr (EPI == 0) {
          C[idx] = c[fm][fn][r];
        } else if constexpr (EPI == 1) {
          float dt = c[fm][fn][r] + aux[n];
          float dc = 0.5f + 0.5f * dt / (1.0f + fabsf(dt));
          C[idx] = rintf(dc * 32768.0f) * (1.0f / 32768.0f);
        } else {
          C[idx] = c[fm][fn][r] + aux[idx];
        }
      }
}

// -------------------------------------- skinny GEMM (N=64, direct global) ----
__global__ __launch_bounds__(128) void gemm_dbl_kernel(const _Float16* __restrict__ A,
                                                       const _Float16* __restrict__ B,
                                                       _Float16* __restrict__ C,
                                                       int N, int K) {
  const int lane = threadIdx.x & 31, w = threadIdx.x >> 5;
  const int m0 = blockIdx.y * 64 + (w & 1) * 32;
  const int n0 = (w >> 1) * 32;
  v8f c[2][2] = {};
  for (int k = 0; k < K; k += 32) {
    v16h a0 = load_frag_a(A, K, m0,      k, lane);
    v16h a1 = load_frag_a(A, K, m0 + 16, k, lane);
    v16h b0 = load_frag_b_ptr(B + (long)(n0      + (lane & 15)) * K + k + ((lane & 16) ? 8 : 0));
    v16h b1 = load_frag_b_ptr(B + (long)(n0 + 16 + (lane & 15)) * K + k + ((lane & 16) ? 8 : 0));
    c[0][0] = WMMA_F16(a0, b0, c[0][0]);
    c[0][1] = WMMA_F16(a0, b1, c[0][1]);
    c[1][0] = WMMA_F16(a1, b0, c[1][0]);
    c[1][1] = WMMA_F16(a1, b1, c[1][1]);
  }
  const int rb = (lane & 16) ? 8 : 0, nc = lane & 15;
  #pragma unroll
  for (int fm = 0; fm < 2; ++fm)
    #pragma unroll
    for (int fn = 0; fn < 2; ++fn)
      #pragma unroll
      for (int r = 0; r < 8; ++r)
        C[(long)(m0 + fm * 16 + rb + r) * N + (n0 + fn * 16 + nc)] = (_Float16)c[fm][fn][r];
}

// ------------------------------------------------------------- reductions ----

__global__ __launch_bounds__(256) void absum_partial_kernel(const float* __restrict__ W,
                                                            long n, float* __restrict__ part) {
  __shared__ float sm[256];
  float acc = 0.0f;
  for (long i = (long)blockIdx.x * 256 + threadIdx.x; i < n; i += (long)gridDim.x * 256)
    acc += fabsf(W[i]);
  sm[threadIdx.x] = acc;
  __syncthreads();
  for (int s = 128; s > 0; s >>= 1) {
    if ((int)threadIdx.x < s) sm[threadIdx.x] += sm[threadIdx.x + s];
    __syncthreads();
  }
  if (threadIdx.x == 0) part[blockIdx.x] = sm[0];
}

__global__ __launch_bounds__(64) void absum_final_kernel(const float* __restrict__ part,
                                                         float inv_n, float* __restrict__ s_out) {
  __shared__ float sm[64];
  sm[threadIdx.x] = part[threadIdx.x];
  __syncthreads();
  for (int s = 32; s > 0; s >>= 1) {
    if ((int)threadIdx.x < s) sm[threadIdx.x] += sm[threadIdx.x + s];
    __syncthreads();
  }
  if (threadIdx.x == 0) *s_out = sm[0] * inv_n + 1e-8f;  // s = mean|W| + 1e-8
}

__global__ __launch_bounds__(256) void quantize_kernel(const float* __restrict__ W, long n,
                                                       const float* __restrict__ s_ptr,
                                                       _Float16* __restrict__ Wq) {
  long i = (long)blockIdx.x * 256 + threadIdx.x;
  if (i >= n) return;
  float s = *s_ptr;
  float q = fminf(fmaxf(rintf(W[i] / s), -1.0f), 1.0f) * s;  // ternary {-s,0,s}: exact in f16
  Wq[i] = (_Float16)q;
}

// ---------------------------------------------------------------- norm -------

__global__ __launch_bounds__(256) void norm_kernel(const float* __restrict__ x,
                                                   const float* __restrict__ gamma,
                                                   const float* __restrict__ step,
                                                   _Float16* __restrict__ xn) {
  __shared__ float sm[256];
  __shared__ float mean_s, inv_s;
  const int t = blockIdx.x;
  const float* row = x + (long)t * DMODEL;
  float s0 = 0.0f;
  #pragma unroll
  for (int j = 0; j < 4; ++j) s0 += row[threadIdx.x + j * 256];
  sm[threadIdx.x] = s0;
  __syncthreads();
  for (int s = 128; s > 0; s >>= 1) {
    if ((int)threadIdx.x < s) sm[threadIdx.x] += sm[threadIdx.x + s];
    __syncthreads();
  }
  if (threadIdx.x == 0) mean_s = sm[0] * (1.0f / DMODEL);
  __syncthreads();
  float mean = mean_s;
  float s1 = 0.0f;
  #pragma unroll
  for (int j = 0; j < 4; ++j) {
    float d = row[threadIdx.x + j * 256] - mean;
    s1 += d * d;
  }
  sm[threadIdx.x] = s1;
  __syncthreads();
  for (int s = 128; s > 0; s >>= 1) {
    if ((int)threadIdx.x < s) sm[threadIdx.x] += sm[threadIdx.x + s];
    __syncthreads();
  }
  if (threadIdx.x == 0) inv_s = pow2scale(sm[0] * (1.0f / DMODEL));  // STE forward value
  __syncthreads();
  float g = inv_s * step[0];
  #pragma unroll
  for (int j = 0; j < 4; ++j) {
    int i = threadIdx.x + j * 256;
    xn[(long)t * DMODEL + i] = (_Float16)((row[i] - mean) * g * gamma[i]);
  }
}

// ----------------------------------------------- conv(4, causal) + squareplus

__global__ __launch_bounds__(256) void conv_sp_kernel(const float* __restrict__ xz,
                                                      const float* __restrict__ conv_w,
                                                      const float* __restrict__ conv_b,
                                                      _Float16* __restrict__ xcs) {
  long idx = (long)blockIdx.x * 256 + threadIdx.x;
  if (idx >= (long)NTOK * DINNER) return;
  int d = (int)(idx % DINNER);
  long bt = idx / DINNER;
  int t = (int)(bt % LSEQ);
  long b = bt / LSEQ;
  const float* wk = conv_w + (long)d * 4;
  float acc = conv_b[d];
  #pragma unroll
  for (int j = 0; j < 4; ++j) {
    int tt = t + j - 3;
    if (tt >= 0) acc += wk[j] * xz[((b * LSEQ + tt) * (long)XZCOLS) + d];
  }
  xcs[idx] = (_Float16)squareplus(acc);
}

// ----------------------------------------------- sequential scan + gate ------

__global__ __launch_bounds__(256) void scan_gate_kernel(const float* __restrict__ decay,
                                                        const _Float16* __restrict__ u,
                                                        const float* __restrict__ xz,
                                                        _Float16* __restrict__ y) {
  int gid = blockIdx.x * 256 + threadIdx.x;  // [0, NBATCH*DINNER)
  int d = gid % DINNER;
  int b = gid / DINNER;
  long base = (long)b * LSEQ * DINNER + d;
  long zbase = (long)b * LSEQ * XZCOLS + DINNER + d;
  float h = 0.0f;
  for (int t = 0; t < LSEQ; ++t) {
    float a = decay[base + (long)t * DINNER];
    float uu = (float)u[base + (long)t * DINNER];
    h = a * h + (1.0f - a) * uu;
    float z = xz[zbase + (long)t * XZCOLS];
    y[base + (long)t * DINNER] = (_Float16)(h * squareplus(z));
  }
}

// ---------------------------------------------------------------- launch -----

extern "C" void kernel_launch(void* const* d_in, const int* in_sizes, int n_in,
                              void* d_out, int out_size, void* d_ws, size_t ws_size,
                              hipStream_t stream) {
  const float* x      = (const float*)d_in[0];
  const float* gamma  = (const float*)d_in[1];
  const float* step   = (const float*)d_in[2];
  const float* W_in   = (const float*)d_in[3];
  const float* conv_w = (const float*)d_in[4];
  const float* conv_b = (const float*)d_in[5];
  const float* W_x    = (const float*)d_in[6];
  const float* W_dt   = (const float*)d_in[7];
  const float* b_dt   = (const float*)d_in[8];
  const float* W_out  = (const float*)d_in[9];
  float* out = (float*)d_out;

  char* w = (char*)d_ws;
  size_t off = 0;
  auto take = [&](size_t bytes) -> void* {
    void* p = w + off;
    off += (bytes + 255) & ~(size_t)255;
    return p;
  };
  float*    scl    = (float*)    take(4 * sizeof(float));
  float*    part   = (float*)    take(4 * 64 * sizeof(float));
  _Float16* Wq_in  = (_Float16*) take((size_t)4096 * 1024 * 2);
  _Float16* Wq_x   = (_Float16*) take((size_t)64 * 2048 * 2);   // rows 0..63 only
  _Float16* Wq_dt  = (_Float16*) take((size_t)2048 * 64 * 2);
  _Float16* Wq_out = (_Float16*) take((size_t)1024 * 2048 * 2);
  _Float16* xn     = (_Float16*) take((size_t)NTOK * DMODEL * 2);
  float*    xz     = (float*)    take((size_t)NTOK * XZCOLS * 4);
  _Float16* xcs    = (_Float16*) take((size_t)NTOK * DINNER * 2);
  _Float16* dbl64  = (_Float16*) take((size_t)NTOK * DTRANK * 2);
  float*    decay  = (float*)    take((size_t)NTOK * DINNER * 4);
  _Float16* ybuf   = (_Float16*) take((size_t)NTOK * DINNER * 2);

  // 1) ternary weight scales s = mean|W| + 1e-8 (deterministic two-stage reduce)
  const float* Wm[4] = {W_in, W_x, W_dt, W_out};
  const long   nW[4] = {4096L * 1024, 96L * 2048, 2048L * 64, 1024L * 2048};
  for (int i = 0; i < 4; ++i) {
    absum_partial_kernel<<<64, 256, 0, stream>>>(Wm[i], nW[i], part + i * 64);
    absum_final_kernel<<<1, 64, 0, stream>>>(part + i * 64, (float)(1.0 / (double)nW[i]), scl + i);
  }
  // 2) quantize to f16 ternary
  quantize_kernel<<<(unsigned)((4096L * 1024 + 255) / 256), 256, 0, stream>>>(W_in, 4096L * 1024, scl + 0, Wq_in);
  quantize_kernel<<<(unsigned)((64L * 2048 + 255) / 256), 256, 0, stream>>>(W_x, 64L * 2048, scl + 1, Wq_x);
  quantize_kernel<<<(unsigned)((2048L * 64 + 255) / 256), 256, 0, stream>>>(W_dt, 2048L * 64, scl + 2, Wq_dt);
  quantize_kernel<<<(unsigned)((1024L * 2048 + 255) / 256), 256, 0, stream>>>(W_out, 1024L * 2048, scl + 3, Wq_out);

  // 3) bitshift norm -> f16 activations
  norm_kernel<<<NTOK, 256, 0, stream>>>(x, gamma, step, xn);

  // 4) xz = xn @ Wq_in^T   [8192 x 4096], K=1024   (TDM-staged B, fused store)
  gemm_tdm_kernel<0><<<dim3(XZCOLS / 128, NTOK / 64), 128, 0, stream>>>(xn, Wq_in, nullptr, xz, XZCOLS, DMODEL);

  // 5) causal depthwise conv(4) + squareplus -> xcs (f16)
  conv_sp_kernel<<<(unsigned)(((long)NTOK * DINNER + 255) / 256), 256, 0, stream>>>(xz, conv_w, conv_b, xcs);

  // 6) dbl64 = xcs @ Wq_x[:64]^T   [8192 x 64], K=2048
  gemm_dbl_kernel<<<dim3(1, NTOK / 64), 128, 0, stream>>>(xcs, Wq_x, dbl64, DTRANK, DINNER);

  // 7) decay = round(sigmoid(dbl64 @ Wq_dt^T + b_dt) * 2^15)/2^15  [8192 x 2048], K=64
  gemm_tdm_kernel<1><<<dim3(DINNER / 128, NTOK / 64), 128, 0, stream>>>(dbl64, Wq_dt, b_dt, decay, DINNER, DTRANK);

  // 8) dyadic scan + gate: y = h * squareplus(z)
  scan_gate_kernel<<<(NBATCH * DINNER) / 256, 256, 0, stream>>>(decay, xcs, xz, ybuf);

  // 9) out = y @ Wq_out^T + x   [8192 x 1024], K=2048
  gemm_tdm_kernel<2><<<dim3(DMODEL / 128, NTOK / 64), 128, 0, stream>>>(ybuf, Wq_out, x, out, DMODEL, DINNER);

  (void)in_sizes; (void)n_in; (void)out_size; (void)ws_size;
}
